// SpectralInitializer_25563645346549
// MI455X (gfx1250) — matrix operand order repaired and checked by reading, not compiled, not run on hardware
//
#include <hip/hip_runtime.h>
#include <cstdint>
#include <cstddef>

// ---------------- problem constants (from reference) ----------------
#define B_   8
#define HW_  64
#define D_   768
#define NMAX 1024
#define KNN  20
#define KPS  4        // K_PER_SCALE
#define NPI  50       // NUM_POWER_ITERS

typedef __attribute__((ext_vector_type(16))) __bf16 v16bf;
typedef __attribute__((ext_vector_type(8)))  __bf16 bf16x8;
typedef __attribute__((ext_vector_type(8)))  float  v8f;

// ---------------- small device helpers ----------------
__device__ __forceinline__ uint32_t hash3(uint32_t a, uint32_t b, uint32_t c) {
  uint32_t h = a * 0x9E3779B9u ^ b * 0x85EBCA6Bu ^ c * 0xC2B2AE35u;
  h ^= h >> 16; h *= 0x7FEB352Du;
  h ^= h >> 15; h *= 0x846CA68Bu;
  h ^= h >> 16;
  return h;
}
__device__ __forceinline__ float u01(uint32_t h) {
  return (float)(h >> 8) * (1.0f / 16777216.0f);
}
__device__ __forceinline__ float gaussf(uint32_t a, uint32_t b, uint32_t c) {
  float u1 = fmaxf(u01(hash3(a, b, c)), 1e-7f);
  float u2 = u01(hash3(a ^ 0x5bd1e995u, b, c));
  return sqrtf(-2.0f * logf(u1)) * cosf(6.28318530718f * u2);
}
__device__ __forceinline__ float block_reduce_sum(float x, float* sbuf, int n) {
  int t = threadIdx.x;
  sbuf[t] = x; __syncthreads();
  for (int st = n >> 1; st > 0; st >>= 1) {
    if (t < st) sbuf[t] += sbuf[t + st];
    __syncthreads();
  }
  float r = sbuf[0];
  __syncthreads();
  return r;
}

// ---------------- 1) adaptive average pool (+ f32 row norms + bf16 copy) ----
__global__ void pool_kernel(const float* __restrict__ feats,
                            float* __restrict__ pooled,
                            unsigned short* __restrict__ pooledh,
                            float* __restrict__ sq, int s) {
  int n = blockIdx.x;          // pooled cell index (0..N-1)
  int b = blockIdx.y;
  int N = s * s;
  int r = HW_ / s;
  int cy = n / s, cx = n % s;
  int y0 = cy * r, x0 = cx * r;
  float inv = 1.0f / (float)(r * r);
  __bf16* ph = (__bf16*)pooledh;
  __shared__ float sbuf[256];
  float part = 0.f;
  for (int d = threadIdx.x; d < D_; d += 256) {
    float acc = 0.f;
    for (int py = 0; py < r; ++py) {
      const float* row = feats + (((size_t)b * HW_ + (y0 + py)) * HW_ + x0) * D_ + d;
      for (int px = 0; px < r; ++px) acc += row[(size_t)px * D_];
    }
    float m = acc * inv;
    size_t o = ((size_t)b * N + n) * D_ + d;
    pooled[o] = m;
    ph[o] = (__bf16)m;
    part += m * m;
  }
  sbuf[threadIdx.x] = part; __syncthreads();
  for (int st = 128; st > 0; st >>= 1) {
    if (threadIdx.x < st) sbuf[threadIdx.x] += sbuf[threadIdx.x + st];
    __syncthreads();
  }
  if (threadIdx.x == 0) sq[b * N + n] = sbuf[0];
}

// ---------------- 2) WMMA Gram matrix -> pairwise distance matrix ----------
// one wave32 per 16x16 output tile; K-loop of v_wmma_f32_16x16x32_bf16
__global__ void __launch_bounds__(32)
gram_dist_kernel(const unsigned short* __restrict__ pooledh,
                 const float* __restrict__ sq,
                 float* __restrict__ dist, int N) {
  int lane = threadIdx.x;
  int m = lane & 15, g = lane >> 4;
  int bx = blockIdx.x, by = blockIdx.y, b = blockIdx.z;
  const __bf16* F  = (const __bf16*)pooledh + (size_t)b * N * D_;
  const __bf16* Fa = F + (size_t)(bx * 16 + m) * D_;   // A rows (M dim)
  const __bf16* Fb = F + (size_t)(by * 16 + m) * D_;   // B cols (N dim) = rows of F
  v8f acc = {};
  for (int k0 = 0; k0 < D_; k0 += 32) {
    __builtin_prefetch(Fa + k0 + 128, 0, 1);
    __builtin_prefetch(Fb + k0 + 128, 0, 1);
    // A 16x32 bf16: lane(m,g) holds K = [k0+16g .. +15]
    bf16x8 alo = *(const bf16x8*)(Fa + k0 + 16 * g);
    bf16x8 ahi = *(const bf16x8*)(Fa + k0 + 16 * g + 8);
    v16bf a;
#pragma unroll
    for (int i = 0; i < 8; ++i) { a[i] = alo[i]; a[i + 8] = ahi[i]; }
    // B 32x16 bf16: lane(n=m,g) holds K = [k0+16g .. +15]
    v16bf bm = *(const v16bf*)(Fb + k0 + 16 * g);
    acc = __builtin_amdgcn_wmma_f32_16x16x32_bf16(
        /*neg_a=*/false, a, /*neg_b=*/false, bm,
        /*c_mod=*/(short)0, acc, /*reuse_a=*/false, /*reuse_b=*/false);
  }
  const float* sqb = sq + b * N;
#pragma unroll
  for (int r0 = 0; r0 < 8; ++r0) {
    int gi = bx * 16 + 8 * g + r0;   // D row  (M)
    int gj = by * 16 + m;            // D col  (N)
    float d2 = sqb[gi] + sqb[gj] - 2.0f * acc[r0];
    dist[((size_t)b * N + gi) * N + gj] = sqrtf(fmaxf(d2, 0.0f));
  }
}

// ---------------- 3) per-row kNN selection -> half-adjacency (in place) ----
__global__ void __launch_bounds__(32)
knn_row_kernel(float* __restrict__ dist, int N) {
  int i = blockIdx.x, b = blockIdx.y;
  int lane = threadIdx.x;
  float* row = dist + ((size_t)b * N + i) * N;
  int cnt = N >> 5;                      // elements per lane (<=32)
  float rv[32];
  for (int q = 0; q < cnt; ++q) rv[q] = row[lane + 32 * q];
  int ci[KNN]; float cw[KNN];
  for (int t = 0; t <= KNN; ++t) {       // round 0 removes self (d==0)
    float bv = 3.4e38f; int bq = 0;
    for (int q = 0; q < cnt; ++q) if (rv[q] < bv) { bv = rv[q]; bq = q; }
    int bi = lane + 32 * bq;
    for (int off = 16; off > 0; off >>= 1) {    // wave32 argmin
      float ov = __shfl_xor(bv, off);
      int   oi = __shfl_xor(bi, off);
      if (ov < bv || (ov == bv && oi < bi)) { bv = ov; bi = oi; }
    }
    if ((bi & 31) == lane) rv[bi >> 5] = 3.4e38f;   // mask winner
    if (t > 0) { ci[t - 1] = bi; cw[t - 1] = expf(-bv * 0.5f); } // sigma=1
  }
  // rewrite row as sparse-in-dense half adjacency (race-free: each lane owns j%32==lane)
  for (int q = 0; q < cnt; ++q) {
    int j = lane + 32 * q;
    float w = 0.f;
#pragma unroll
    for (int t = 0; t < KNN; ++t) if (ci[t] == j) w = cw[t];
    row[j] = w;
  }
}

// ---------------- 4) symmetrize + degree -> dinv ---------------------------
__global__ void symdeg_kernel(const float* __restrict__ ah,
                              float* __restrict__ wsym,
                              float* __restrict__ dinv, int N) {
  int i = blockIdx.x, b = blockIdx.y;
  const float* base = ah + (size_t)b * N * N;
  float* wrow = wsym + ((size_t)b * N + i) * N;
  __shared__ float sbuf[256];
  float acc = 0.f;
  for (int j = threadIdx.x; j < N; j += 256) {
    float w = 0.5f * (base[(size_t)i * N + j] + base[(size_t)j * N + i]);
    wrow[j] = w;
    acc += w;
  }
  sbuf[threadIdx.x] = acc; __syncthreads();
  for (int st = 128; st > 0; st >>= 1) {
    if (threadIdx.x < st) sbuf[threadIdx.x] += sbuf[threadIdx.x + st];
    __syncthreads();
  }
  if (threadIdx.x == 0) dinv[b * N + i] = rsqrtf(sbuf[0] + 1e-8f);
}

// ---------------- 5) Wn = dinv_i * W * dinv_j (in place) -------------------
__global__ void normW_kernel(float* __restrict__ wsym,
                             const float* __restrict__ dinv, int N) {
  int i = blockIdx.x, b = blockIdx.y;
  float di = dinv[b * N + i];
  float* wrow = wsym + ((size_t)b * N + i) * N;
  const float* dv = dinv + b * N;
  for (int j = threadIdx.x; j < N; j += 256) wrow[j] *= di * dv[j];
}

// ---------------- 6) deflated power iteration, one workgroup per image -----
__global__ void power_kernel(const float* __restrict__ Wn,
                             const float* __restrict__ pooled,
                             float* __restrict__ eig, int N) {
  int b = blockIdx.x, tid = threadIdx.x;     // blockDim.x == N
  __shared__ float v[NMAX];
  __shared__ float red[NMAX];
  const float* wrow = Wn + ((size_t)b * N + tid) * N;
  float evloc[KPS];
  for (int e = 0; e < KPS; ++e) {
    float x;
    if (e == 0) {
      const float* fr = pooled + ((size_t)b * N + tid) * D_;
      float a = 0.f;
      for (int d = 0; d < D_; ++d) a += fr[d];
      x = a * (1.0f / (float)D_);            // feats.mean(axis=1)
    } else {
      x = gaussf((uint32_t)(b * 131 + 7), (uint32_t)e, (uint32_t)tid);
    }
    float nrm = sqrtf(block_reduce_sum(x * x, red, N));
    x = x / (nrm + 1e-8f);
    v[tid] = x; __syncthreads();
    for (int it = 0; it < NPI; ++it) {
      float acc = 0.f;
      for (int j = 0; j < N; ++j) acc = fmaf(wrow[j], v[j], acc);   // Wn @ v
#pragma unroll
      for (int p = 0; p < KPS - 1; ++p) {                           // deflation
        if (p < e) {
          float dp = block_reduce_sum(acc * evloc[p], red, N);
          acc -= dp * evloc[p];
        }
      }
      float nn = sqrtf(block_reduce_sum(acc * acc, red, N));
      acc = acc / (nn + 1e-8f);
      float df = v[tid] - acc;
      float dn = sqrtf(block_reduce_sum(df * df, red, N));
      if (dn < 1e-6f) break;                 // uniform across block
      v[tid] = acc; __syncthreads();
    }
    evloc[e] = v[tid];
    eig[((size_t)b * KPS + e) * N + tid] = v[tid];
    __syncthreads();
  }
}

// ---------------- 7) kmeans++ in combined [2*eig | fn] space ---------------
__device__ float comb_dist(const float* __restrict__ F, const float* __restrict__ E,
                           const float* __restrict__ sqb, int N, int i, int c) {
  float acc = 0.f;
#pragma unroll
  for (int p = 0; p < KPS; ++p) {
    float de = 2.0f * (E[(size_t)p * N + i] - E[(size_t)p * N + c]);
    acc += de * de;
  }
  float ri = 1.0f / fmaxf(sqrtf(sqb[i]), 1e-12f);
  float rc = 1.0f / fmaxf(sqrtf(sqb[c]), 1e-12f);
  const float* fi = F + (size_t)i * D_;
  const float* fc = F + (size_t)c * D_;
  float s = 0.f;
  for (int d = 0; d < D_; ++d) {
    float df = fi[d] * ri - fc[d] * rc;
    s = fmaf(df, df, s);
  }
  return sqrtf(acc + s);
}

__global__ void kmeanspp_kernel(const float* __restrict__ pooled,
                                const float* __restrict__ sq,
                                const float* __restrict__ eig,
                                float* __restrict__ out, int N, int si) {
  int b = blockIdx.x, tid = threadIdx.x;     // 256 threads
  __shared__ float mind[NMAX];
  __shared__ float stot[256];
  __shared__ int   cent[KPS];
  const float* F   = pooled + (size_t)b * N * D_;
  const float* sqb = sq + b * N;
  const float* E   = eig + (size_t)b * KPS * N;
  if (tid == 0) cent[0] = (int)(hash3(0x9999u, (uint32_t)b, 0x1234u) % (uint32_t)N);
  __syncthreads();
  int c0 = cent[0];
  for (int i = tid; i < N; i += 256) mind[i] = comb_dist(F, E, sqb, N, i, c0);
  __syncthreads();
  for (int t = 1; t < KPS; ++t) {
    float part = 0.f;
    for (int i = tid; i < N; i += 256) { float m = mind[i]; part += m * m; }
    stot[tid] = part; __syncthreads();
    for (int st = 128; st > 0; st >>= 1) {
      if (tid < st) stot[tid] += stot[tid + st];
      __syncthreads();
    }
    if (tid == 0) {
      float tot = stot[0] + 1e-8f;
      float u = u01(hash3(0xABCDu, (uint32_t)b, (uint32_t)t)) * tot;
      float cum = 0.f; int sel = N - 1;
      for (int j = 0; j < N; ++j) {
        float m = mind[j]; cum += m * m;
        if (cum >= u) { sel = j; break; }
      }
      cent[t] = sel;
    }
    __syncthreads();
    int c = cent[t];
    for (int i = tid; i < N; i += 256) {
      float dn = comb_dist(F, E, sqb, N, i, c);
      mind[i] = fminf(mind[i], dn);
    }
    __syncthreads();
  }
  // gather slot features (pooled f32 rows) into output (B, 12, 768)
  for (int t = 0; t < KPS; ++t) {
    const float* src = F + (size_t)cent[t] * D_;
    float* dst = out + (((size_t)b * 12) + (size_t)si * KPS + t) * D_;
    for (int d = tid; d < D_; d += 256) dst[d] = src[d];
  }
}

// ---------------- launch ---------------------------------------------------
extern "C" void kernel_launch(void* const* d_in, const int* in_sizes, int n_in,
                              void* d_out, int out_size, void* d_ws, size_t ws_size,
                              hipStream_t stream) {
  (void)in_sizes; (void)n_in; (void)out_size; (void)ws_size;
  const float* feats = (const float*)d_in[0];
  float* out = (float*)d_out;

  char* ws = (char*)d_ws;
  size_t off = 0;
  float*          pooled  = (float*)(ws + off);          off += (size_t)B_ * NMAX * D_ * 4;     // 24 MB
  unsigned short* pooledh = (unsigned short*)(ws + off); off += (size_t)B_ * NMAX * D_ * 2;     // 12 MB
  float*          sq      = (float*)(ws + off);          off += (size_t)B_ * NMAX * 4;          // 32 KB
  float*          dist    = (float*)(ws + off);          off += (size_t)B_ * NMAX * NMAX * 4;   // 32 MB
  float*          wsym    = (float*)(ws + off);          off += (size_t)B_ * NMAX * NMAX * 4;   // 32 MB
  float*          dinv    = (float*)(ws + off);          off += (size_t)B_ * NMAX * 4;          // 32 KB
  float*          eig     = (float*)(ws + off);          off += (size_t)B_ * NMAX * KPS * 4;    // 128 KB

  const int scales[3] = {8, 16, 32};
  for (int si = 0; si < 3; ++si) {
    int s = scales[si];
    int N = s * s;
    pool_kernel     <<<dim3(N, B_),            256, 0, stream>>>(feats, pooled, pooledh, sq, s);
    gram_dist_kernel<<<dim3(N / 16, N / 16, B_), 32, 0, stream>>>(pooledh, sq, dist, N);
    knn_row_kernel  <<<dim3(N, B_),             32, 0, stream>>>(dist, N);
    symdeg_kernel   <<<dim3(N, B_),            256, 0, stream>>>(dist, wsym, dinv, N);
    normW_kernel    <<<dim3(N, B_),            256, 0, stream>>>(wsym, dinv, N);
    power_kernel    <<<dim3(B_),                 N, 0, stream>>>(wsym, pooled, eig, N);
    kmeanspp_kernel <<<dim3(B_),               256, 0, stream>>>(pooled, sq, eig, out, N, si);
  }
}
